// DualAttention_87531433493062
// MI455X (gfx1250) — matrix operand order repaired and checked by reading, not compiled, not run on hardware
//
#include <hip/hip_runtime.h>

#define B_ 64
#define L_ 256
#define D_ 256
#define M_ 512
#define NIT_ 50

typedef __attribute__((ext_vector_type(16))) _Float16 v16h;
typedef __attribute__((ext_vector_type(8)))  _Float16 v8h;
typedef __attribute__((ext_vector_type(8)))  float    v8f;

__device__ __forceinline__ float fexp2(float x) { return __builtin_amdgcn_exp2f(x); }   // v_exp_f32
__device__ __forceinline__ float flog2(float x) { return __builtin_amdgcn_logf(x); }    // v_log_f32
__device__ __forceinline__ float fsigmoid(float s) {
  // 1/(1+exp(-s)) with exp(-s) = exp2(-s*log2(e))
  return 1.f / (1.f + fexp2(-s * 1.44269504088896340736f));
}

__device__ __forceinline__ v8f wmma16(v16h a, v16h b, v8f c) {
  return __builtin_amdgcn_wmma_f32_16x16x32_f16(false, a, false, b, (short)0, c, false, false);
}

// A fragment: 16x32 f16, row = lane&15, halves 0..7 -> K=kh.., halves 8..15 -> K=16+kh..
__device__ __forceinline__ v16h load_fragA(const _Float16* base, int ld, int k0) {
  int lane = threadIdx.x & 31;
  const _Float16* p = base + (size_t)(lane & 15) * ld + k0 + ((lane >> 4) << 3);
  union { v16h v; v8h h[2]; } u;
  u.h[0] = *(const v8h*)(p);
  u.h[1] = *(const v8h*)(p + 16);
  return u.v;
}

// B fragment: 32x16 f16, col = lane&15, halves 0..7 -> K=kh.., halves 8..15 -> K=16+kh..
__device__ __forceinline__ v16h load_fragB(const _Float16* base, int ld, int col0, int k0) {
  int lane = threadIdx.x & 31;
  const _Float16* p = base + (size_t)(col0 + (lane & 15)) * ld + k0 + ((lane >> 4) << 3);
  union { v16h v; v8h h[2]; } u;
  u.h[0] = *(const v8h*)(p);
  u.h[1] = *(const v8h*)(p + 16);
  return u.v;
}

// ---------------- prep kernels ----------------
__global__ void k_prep_x(const float* __restrict__ xemb, const float* __restrict__ pos,
                         _Float16* __restrict__ x_h, _Float16* __restrict__ xT_h) {
  size_t idx = (size_t)blockIdx.x * 256 + threadIdx.x;  // B*M*D
  int d = (int)(idx & (D_ - 1));
  int m = (int)((idx >> 8) & (M_ - 1));
  int b = (int)(idx >> 17);
  float v = (m < L_) ? pos[((size_t)b * L_ + m) * D_ + d]
                     : xemb[((size_t)b * L_ + (m - L_)) * D_ + d];
  x_h[idx] = (_Float16)v;
  xT_h[((size_t)b * D_ + d) * M_ + m] = (_Float16)v;
}

__global__ void k_prep_w(const float* __restrict__ wq, const float* __restrict__ w1,
                         const float* __restrict__ w2, const float* __restrict__ wa1,
                         _Float16* __restrict__ wq_h, _Float16* __restrict__ w1_h,
                         _Float16* __restrict__ w2_h, _Float16* __restrict__ wa1T_h) {
  int idx = blockIdx.x * 256 + threadIdx.x;  // D*D
  int j = idx >> 8, d = idx & 255;
  wq_h[idx]   = (_Float16)wq[idx];
  w1_h[idx]   = (_Float16)w1[idx];
  w2_h[idx]   = (_Float16)w2[idx];
  wa1T_h[idx] = (_Float16)wa1[d * D_ + j];  // transpose: element (k=d, n=j)
}

__global__ void k_alpha_ent(const float* __restrict__ xemb, const float* __restrict__ aww,
                            const float* __restrict__ awb, float* __restrict__ a_ent) {
  __shared__ float red[256];
  int b = blockIdx.x, t = threadIdx.x;
  red[t] = xemb[((size_t)b * L_ + (L_ - 1)) * D_ + t] * aww[t];
  __syncthreads();
  for (int s = 128; s > 0; s >>= 1) { if (t < s) red[t] += red[t + s]; __syncthreads(); }
  if (t == 0) {
    float s = red[0] + awb[0];
    float v = fsigmoid(s) + 1.f;
    if (v == 1.f) v = 1.00001f;
    a_ent[b] = v;
  }
}

// ---------------- generic GEMM: Out = relu(A @ W^T + bias), f16 out ----------------
__global__ void __launch_bounds__(256) k_gemm_relu(const _Float16* __restrict__ A,
    const _Float16* __restrict__ W, const float* __restrict__ bias,
    _Float16* __restrict__ Out) {
  int rb = blockIdx.x * 16;
  int w = threadIdx.x >> 5, lane = threadIdx.x & 31;
  int colb = w * 32;
  const _Float16* Ab = A + (size_t)rb * D_;
  v8f c0 = {}, c1 = {};
  for (int k = 0; k < D_; k += 32) {
    v16h a  = load_fragA(Ab, D_, k);
    v16h b0 = load_fragB(W, D_, colb, k);
    v16h b1 = load_fragB(W, D_, colb + 16, k);
    c0 = wmma16(a, b0, c0);
    c1 = wmma16(a, b1, c1);
  }
  int n = lane & 15, mo = (lane >> 4) << 3;
  for (int t = 0; t < 2; ++t) {
    v8f c = t ? c1 : c0;
    int col = colb + t * 16 + n;
    float bv = bias[col];
#pragma unroll
    for (int r = 0; r < 8; ++r) {
      float v = c[r] + bv;
      Out[(size_t)(rb + mo + r) * D_ + col] = (_Float16)(v > 0.f ? v : 0.f);
    }
  }
}

// -------- fused: scores tile -> mask -> entmax bisection -> attn @ x_ --------
__global__ void __launch_bounds__(256) k_attn(const _Float16* __restrict__ q_h,
    const _Float16* __restrict__ x_h, const _Float16* __restrict__ xT_h,
    const int* __restrict__ mask, const float* __restrict__ a_ent,
    float* __restrict__ attv_f, _Float16* __restrict__ attv_h) {
  __shared__ float    sc[16 * M_];      // 32 KB
  __shared__ _Float16 at[16 * M_];      // 16 KB
  int b  = blockIdx.x >> 5;
  int rt = (blockIdx.x & 31) * 16;
  int w = threadIdx.x >> 5, lane = threadIdx.x & 31;
  const _Float16* Aq = q_h + ((size_t)b * M_ + rt) * D_;
  const _Float16* Bx = x_h + (size_t)b * M_ * D_;

  // scores: wave w -> cols [64w, 64w+64)
  v8f s[4] = {};
  for (int k = 0; k < D_; k += 32) {
    v16h a = load_fragA(Aq, D_, k);
#pragma unroll
    for (int t = 0; t < 4; ++t) {
      v16h bf = load_fragB(Bx, D_, w * 64 + t * 16, k);
      s[t] = wmma16(a, bf, s[t]);
    }
  }
  int n = lane & 15, mo = (lane >> 4) << 3;
#pragma unroll
  for (int t = 0; t < 4; ++t) {
    int col = w * 64 + t * 16 + n;
#pragma unroll
    for (int r = 0; r < 8; ++r) sc[(mo + r) * M_ + col] = s[t][r] * 0.0625f;  // /sqrt(256)
  }
  __syncthreads();

  float alpha = a_ent[b];
  float am1 = alpha - 1.f, inv = 1.f / am1;
  for (int idx = threadIdx.x; idx < 16 * M_; idx += 256) {
    int kk = idx & (M_ - 1);
    float v = sc[idx] * am1;
    sc[idx] = mask[b * L_ + (kk & (L_ - 1))] ? v : -__builtin_inff();
  }
  __syncthreads();

  // bisection: 16 lanes per row, 32 elements per lane, cached in VGPRs
  int row = w * 2 + (lane >> 4);
  int k0  = (lane & 15) * 32;
  float x[32];
#pragma unroll
  for (int i = 0; i < 32; ++i) x[i] = sc[row * M_ + k0 + i];
  float mx = -__builtin_inff();
#pragma unroll
  for (int i = 0; i < 32; ++i) mx = fmaxf(mx, x[i]);
#pragma unroll
  for (int mk = 1; mk < 16; mk <<= 1) mx = fmaxf(mx, __shfl_xor(mx, mk, 32));

  float tau = mx - 1.f;
  float dm  = 1.f - powf(1.f / (float)M_, am1);  // tau_hi - tau_lo
  float tm  = tau;
  for (int it = 0; it < NIT_; ++it) {
    dm *= 0.5f;
    tm = tau + dm;
    float ssum = 0.f;
#pragma unroll
    for (int i = 0; i < 32; ++i) {
      float v = x[i] - tm;
      if (v > 0.f) ssum += fexp2(inv * flog2(v));
    }
#pragma unroll
    for (int mk = 1; mk < 16; mk <<= 1) ssum += __shfl_xor(ssum, mk, 32);
    if (ssum >= 1.f) tau = tm;
  }
  float p[32]; float psum = 0.f;
#pragma unroll
  for (int i = 0; i < 32; ++i) {
    float v = x[i] - tm;
    p[i] = (v > 0.f) ? fexp2(inv * flog2(v)) : 0.f;
    psum += p[i];
  }
#pragma unroll
  for (int mk = 1; mk < 16; mk <<= 1) psum += __shfl_xor(psum, mk, 32);
  float rn = 1.f / psum;
#pragma unroll
  for (int i = 0; i < 32; ++i) at[row * M_ + k0 + i] = (_Float16)(p[i] * rn);
  __syncthreads();

  // att_v = attn(16x512) @ x_(512x256), wave w -> cols [32w, 32w+32)
  const _Float16* BxT = xT_h + (size_t)b * D_ * M_;
  v8f d0 = {}, d1 = {};
  for (int k = 0; k < M_; k += 32) {
    v16h a  = load_fragA(at, M_, k);
    v16h b0 = load_fragB(BxT, M_, w * 32, k);
    v16h b1 = load_fragB(BxT, M_, w * 32 + 16, k);
    d0 = wmma16(a, b0, d0);
    d1 = wmma16(a, b1, d1);
  }
#pragma unroll
  for (int t = 0; t < 2; ++t) {
    v8f c = t ? d1 : d0;
    int col = w * 32 + t * 16 + n;
#pragma unroll
    for (int r = 0; r < 8; ++r) {
      size_t gm = (size_t)b * M_ + rt + mo + r;
      float v = c[r];
      attv_f[gm * D_ + col] = v;
      attv_h[gm * D_ + col] = (_Float16)v;
    }
  }
}

// -------- GEMM2 + bias + residual + LayerNorm -> local_c (rotated) + h_h f16 --------
__global__ void __launch_bounds__(256) k_mlp_ln(const _Float16* __restrict__ hid_h,
    const _Float16* __restrict__ W2, const float* __restrict__ b2,
    const float* __restrict__ attv_f, const float* __restrict__ ln_g,
    const float* __restrict__ ln_b, float* __restrict__ out_local,
    _Float16* __restrict__ h_h) {
  __shared__ float hh[16 * D_];
  int rb = blockIdx.x * 16;
  int w = threadIdx.x >> 5, lane = threadIdx.x & 31;
  int colb = w * 32;
  const _Float16* Ab = hid_h + (size_t)rb * D_;
  v8f c0 = {}, c1 = {};
  for (int k = 0; k < D_; k += 32) {
    v16h a   = load_fragA(Ab, D_, k);
    v16h bf0 = load_fragB(W2, D_, colb, k);
    v16h bf1 = load_fragB(W2, D_, colb + 16, k);
    c0 = wmma16(a, bf0, c0);
    c1 = wmma16(a, bf1, c1);
  }
  int n = lane & 15, mo = (lane >> 4) << 3;
  for (int t = 0; t < 2; ++t) {
    v8f c = t ? c1 : c0;
    int col = colb + t * 16 + n;
    float bv = b2[col];
#pragma unroll
    for (int r = 0; r < 8; ++r) {
      size_t gm = (size_t)rb + mo + r;
      hh[(mo + r) * D_ + col] = c[r] + bv + attv_f[gm * D_ + col];
    }
  }
  __syncthreads();

  int row = w * 2 + (lane >> 4);
  int off = (lane & 15) * 16;
  float v[16];
  float s1 = 0.f, s2 = 0.f;
#pragma unroll
  for (int i = 0; i < 16; ++i) { v[i] = hh[row * D_ + off + i]; s1 += v[i]; s2 += v[i] * v[i]; }
#pragma unroll
  for (int mk = 1; mk < 16; mk <<= 1) { s1 += __shfl_xor(s1, mk, 32); s2 += __shfl_xor(s2, mk, 32); }
  float mu = s1 * (1.f / D_);
  float var = s2 * (1.f / D_) - mu * mu;
  float rstd = rsqrtf(var + 1e-5f);
  size_t g = (size_t)rb + row;
  int b = (int)(g >> 9);
  int mrow = (int)(g & (M_ - 1));
  int q = (mrow == M_ - 1) ? 0 : mrow + 1;  // local_c = concat([h[-1:], h[:-1]])
#pragma unroll
  for (int i = 0; i < 16; ++i) {
    int d = off + i;
    float y = (v[i] - mu) * rstd * ln_g[d] + ln_b[d];
    out_local[((size_t)b * M_ + q) * D_ + d] = y;
    h_h[g * D_ + d] = (_Float16)y;
  }
}

// -------- per-batch: mvec = m_s @ atten_w2 + bias; a_glob from m_s --------
__global__ void k_msvec(const float* __restrict__ out_local, const float* __restrict__ w2a,
                        const float* __restrict__ abias, const float* __restrict__ aww,
                        const float* __restrict__ awb, float* __restrict__ mvec,
                        float* __restrict__ a_glob) {
  __shared__ float ms[D_];
  __shared__ float red[256];
  int b = blockIdx.x, t = threadIdx.x;
  float msv = out_local[((size_t)b * M_) * D_ + t];  // local_c[b,0,:] == m_s
  ms[t] = msv;
  red[t] = msv * aww[t];
  __syncthreads();
  for (int s = 128; s > 0; s >>= 1) { if (t < s) red[t] += red[t + s]; __syncthreads(); }
  if (t == 0) {
    float s = red[0] + awb[0];
    float v = fsigmoid(s) + 1.f;
    if (v == 1.f) v = 1.00001f;
    a_glob[b] = v;
  }
  float acc = abias[t];
  for (int d = 0; d < D_; ++d) acc += ms[d] * w2a[d * D_ + t];
  mvec[b * D_ + t] = acc;
}

// -------- gate: g[n] = sum_j relu((x_n @ atten_w1)[n,j] + mvec[j]) * w0[j], masked --------
__global__ void __launch_bounds__(256) k_gate(const _Float16* __restrict__ h_h,
    const _Float16* __restrict__ Wa1T, const float* __restrict__ mvec,
    const float* __restrict__ w0, const int* __restrict__ mask,
    float* __restrict__ gbuf) {
  __shared__ float gpart[8][16];
  int b  = blockIdx.x >> 5;
  int nt = (blockIdx.x & 31) * 16;
  int w = threadIdx.x >> 5, lane = threadIdx.x & 31;
  int colb = w * 32;
  const _Float16* Ab = h_h + ((size_t)b * M_ + nt) * D_;
  v8f c0 = {}, c1 = {};
  for (int k = 0; k < D_; k += 32) {
    v16h a   = load_fragA(Ab, D_, k);
    v16h bf0 = load_fragB(Wa1T, D_, colb, k);
    v16h bf1 = load_fragB(Wa1T, D_, colb + 16, k);
    c0 = wmma16(a, bf0, c0);
    c1 = wmma16(a, bf1, c1);
  }
  int n = lane & 15, mo = (lane >> 4) << 3;
  float rowacc[8];
#pragma unroll
  for (int r = 0; r < 8; ++r) rowacc[r] = 0.f;
  for (int t = 0; t < 2; ++t) {
    v8f c = t ? c1 : c0;
    int col = colb + t * 16 + n;
    float mv = mvec[b * D_ + col];
    float wv = w0[col];
#pragma unroll
    for (int r = 0; r < 8; ++r) {
      float v = c[r] + mv;
      rowacc[r] += (v > 0.f ? v : 0.f) * wv;
    }
  }
#pragma unroll
  for (int r = 0; r < 8; ++r) {
    float v = rowacc[r];
#pragma unroll
    for (int mk = 1; mk < 16; mk <<= 1) v += __shfl_xor(v, mk, 32);
    if ((lane & 15) == 0) gpart[w][mo + r] = v;  // lanes 0 and 16
  }
  __syncthreads();
  if (threadIdx.x < 16) {
    float s = 0.f;
#pragma unroll
    for (int ww = 0; ww < 8; ++ww) s += gpart[ww][threadIdx.x];  // deterministic order
    int nn = nt + threadIdx.x;
    if (nn < M_ - 1)
      gbuf[(size_t)b * M_ + nn] = mask[b * L_ + (nn & (L_ - 1))] ? s : -__builtin_inff();
  }
}

// -------- per-batch: entmax over g (K=511) + global_c = sum_n w[n] * x_s[b,n,:] --------
__global__ void k_global(const float* __restrict__ gbuf, const float* __restrict__ a_glob,
                         const float* __restrict__ xemb, const float* __restrict__ pos,
                         float* __restrict__ out_global) {
  __shared__ float gs[M_];
  __shared__ float red[256];
  int b = blockIdx.x, t = threadIdx.x;
  float alpha = a_glob[b];
  float am1 = alpha - 1.f, inv = 1.f / am1;
  for (int idx = t; idx < M_; idx += 256) {
    float v = (idx < M_ - 1) ? gbuf[(size_t)b * M_ + idx] : -__builtin_inff();
    gs[idx] = v * am1;
  }
  __syncthreads();
  red[t] = fmaxf(gs[t], gs[t + 256]);
  __syncthreads();
  for (int s = 128; s > 0; s >>= 1) { if (t < s) red[t] = fmaxf(red[t], red[t + s]); __syncthreads(); }
  float mx = red[0]; __syncthreads();
  float tau = mx - 1.f;
  float dm = 1.f - powf(1.f / 511.f, am1);
  float tm = tau;
  for (int it = 0; it < NIT_; ++it) {
    dm *= 0.5f;
    tm = tau + dm;
    float v0 = gs[t] - tm, v1 = gs[t + 256] - tm;
    float p0 = (v0 > 0.f) ? fexp2(inv * flog2(v0)) : 0.f;
    float p1 = (v1 > 0.f) ? fexp2(inv * flog2(v1)) : 0.f;
    red[t] = p0 + p1; __syncthreads();
    for (int s = 128; s > 0; s >>= 1) { if (t < s) red[t] += red[t + s]; __syncthreads(); }
    float tot = red[0]; __syncthreads();
    if (tot >= 1.f) tau = tm;
  }
  float v0 = gs[t] - tm, v1 = gs[t + 256] - tm;
  float p0 = (v0 > 0.f) ? fexp2(inv * flog2(v0)) : 0.f;
  float p1 = (v1 > 0.f) ? fexp2(inv * flog2(v1)) : 0.f;
  red[t] = p0 + p1; __syncthreads();
  for (int s = 128; s > 0; s >>= 1) { if (t < s) red[t] += red[t + s]; __syncthreads(); }
  float psum = red[0]; __syncthreads();
  gs[t] = p0 / psum;
  gs[t + 256] = p1 / psum;
  __syncthreads();
  float acc = 0.f;
  for (int nn = 0; nn < M_ - 1; ++nn) {
    float xv = (nn < L_) ? pos[((size_t)b * L_ + nn) * D_ + t]
                         : xemb[((size_t)b * L_ + (nn - L_)) * D_ + t];
    acc += gs[nn] * xv;
  }
  out_global[(size_t)b * D_ + t] = acc;
}

extern "C" void kernel_launch(void* const* d_in, const int* in_sizes, int n_in,
                              void* d_out, int out_size, void* d_ws, size_t ws_size,
                              hipStream_t stream) {
  (void)in_sizes; (void)n_in; (void)out_size; (void)ws_size;
  const float* xemb = (const float*)d_in[0];
  const float* pos  = (const float*)d_in[1];
  const int*   mask = (const int*)d_in[2];
  const float* wq   = (const float*)d_in[3];
  const float* bq   = (const float*)d_in[4];
  const float* w1   = (const float*)d_in[5];
  const float* b1   = (const float*)d_in[6];
  const float* w2   = (const float*)d_in[7];
  const float* b2   = (const float*)d_in[8];
  const float* lng  = (const float*)d_in[9];
  const float* lnb  = (const float*)d_in[10];
  const float* aw0  = (const float*)d_in[11];
  const float* aw1  = (const float*)d_in[12];
  const float* aw2  = (const float*)d_in[13];
  const float* ab   = (const float*)d_in[14];
  const float* aww  = (const float*)d_in[15];
  const float* awb  = (const float*)d_in[16];
  float* out_global = (float*)d_out;
  float* out_local  = (float*)d_out + (size_t)B_ * D_;

  char* ws = (char*)d_ws;
  size_t nMD = (size_t)B_ * M_ * D_;
  _Float16* x_h    = (_Float16*)ws; ws += nMD * 2;
  _Float16* xT_h   = (_Float16*)ws; ws += nMD * 2;
  _Float16* q_h    = (_Float16*)ws; ws += nMD * 2;
  _Float16* attv_h = (_Float16*)ws; ws += nMD * 2;
  _Float16* hid_h  = (_Float16*)ws; ws += nMD * 2;
  _Float16* h_h    = (_Float16*)ws; ws += nMD * 2;
  float*    attv_f = (float*)ws;    ws += nMD * 4;
  _Float16* wq_h   = (_Float16*)ws; ws += (size_t)D_ * D_ * 2;
  _Float16* w1_h   = (_Float16*)ws; ws += (size_t)D_ * D_ * 2;
  _Float16* w2_h   = (_Float16*)ws; ws += (size_t)D_ * D_ * 2;
  _Float16* wa1T_h = (_Float16*)ws; ws += (size_t)D_ * D_ * 2;
  float* mvec      = (float*)ws;    ws += (size_t)B_ * D_ * 4;
  float* gbuf      = (float*)ws;    ws += (size_t)B_ * M_ * 4;
  float* a_ent     = (float*)ws;    ws += B_ * 4;
  float* a_glob    = (float*)ws;    ws += B_ * 4;

  k_prep_x<<<(B_ * M_ * D_) / 256, 256, 0, stream>>>(xemb, pos, x_h, xT_h);
  k_prep_w<<<(D_ * D_) / 256, 256, 0, stream>>>(wq, w1, w2, aw1, wq_h, w1_h, w2_h, wa1T_h);
  k_alpha_ent<<<B_, 256, 0, stream>>>(xemb, aww, awb, a_ent);
  k_gemm_relu<<<(B_ * M_) / 16, 256, 0, stream>>>(x_h, wq_h, bq, q_h);
  k_attn<<<B_ * (M_ / 16), 256, 0, stream>>>(q_h, x_h, xT_h, mask, a_ent, attv_f, attv_h);
  k_gemm_relu<<<(B_ * M_) / 16, 256, 0, stream>>>(attv_h, w1_h, b1, hid_h);
  k_mlp_ln<<<(B_ * M_) / 16, 256, 0, stream>>>(hid_h, w2_h, b2, attv_f, lng, lnb, out_local, h_h);
  k_msvec<<<B_, 256, 0, stream>>>(out_local, aw2, ab, aww, awb, mvec, a_glob);
  k_gate<<<B_ * (M_ / 16), 256, 0, stream>>>(h_h, wa1T_h, mvec, aw0, mask, gbuf);
  k_global<<<B_, 256, 0, stream>>>(gbuf, a_glob, xemb, pos, out_global);
}
